// EsmFoldAttention_86827058856663
// MI455X (gfx1250) — matrix operand order repaired and checked by reading, not compiled, not run on hardware
//
#include <hip/hip_runtime.h>
#include <hip/hip_bf16.h>

typedef __attribute__((ext_vector_type(16))) _Float16 v16h;
typedef __attribute__((ext_vector_type(8)))  _Float16 v8h;
typedef __attribute__((ext_vector_type(8)))  float    v8f;

#define BB 8
#define QQ 1024
#define KK 1024
#define CC 256
#define HH 8
#define DH 32
#define HD 256          // H*DH
#define NROW 8192       // B*Q == B*K

__device__ __forceinline__ void wait_ds0() {
    asm volatile("s_wait_dscnt 0" ::: "memory");
}
__device__ __forceinline__ void wait_async_le4() {
    asm volatile("s_wait_asynccnt 0x4" ::: "memory");
}
__device__ __forceinline__ void wait_async_le0() {
    asm volatile("s_wait_asynccnt 0x0" ::: "memory");
}
// CDNA5 async copy: global -> LDS, 16 bytes, tracked by ASYNCcnt.
__device__ __forceinline__ void async_ld_b128(uint32_t lds_addr, const void* gaddr) {
    asm volatile("global_load_async_to_lds_b128 %0, %1, off"
                 :: "v"(lds_addr), "v"(gaddr) : "memory");
}

__device__ __forceinline__ v16h cat8(v8h lo, v8h hi) {
    return __builtin_shufflevector(lo, hi, 0,1,2,3,4,5,6,7,8,9,10,11,12,13,14,15);
}

// A-operand (16x32 f16) lane fragment from f32 row-major source.
__device__ __forceinline__ v16h ldA_f32(const float* __restrict__ p) {
    v16h r;
#pragma unroll
    for (int i = 0; i < 8; ++i) { r[i] = (_Float16)p[i]; r[8 + i] = (_Float16)p[i + 8]; }
    return r;
}

// ---------------------------------------------------------------------------
// Kernel 0: transpose + f32->f16 the five 256x256 weight matrices.
// WT[n][k] = W[k][n], so B-operand fragments become contiguous 32B loads.
// grid = (16,16,5), block = (16,16)
// ---------------------------------------------------------------------------
__global__ __launch_bounds__(256) void esmattn_prep_weights(
    const float* __restrict__ Wq, const float* __restrict__ Wk,
    const float* __restrict__ Wv, const float* __restrict__ Wg,
    const float* __restrict__ Wo, _Float16* __restrict__ WT)  // 5 x 256 x 256
{
    __shared__ float t[16][17];
    const int z = blockIdx.z;
    const float* W = (z == 0) ? Wq : (z == 1) ? Wk : (z == 2) ? Wv
                   : (z == 3) ? Wg : Wo;
    const int k0 = blockIdx.x * 16, n0 = blockIdx.y * 16;
    const int tx = threadIdx.x, ty = threadIdx.y;
    t[ty][tx] = W[(size_t)(k0 + ty) * CC + n0 + tx];   // coalesced along n
    __syncthreads();
    WT[(size_t)z * CC * CC + (size_t)(n0 + ty) * CC + k0 + tx] =
        (_Float16)t[tx][ty];                           // coalesced along k
}

// ---------------------------------------------------------------------------
// Kernel 1: fused Q/K/V/G projections.  X(8192x256) @ W(256x256) in f16 WMMA.
// grid = (NROW/64, HD/16, 4), block = 128 (4 waves, one 16x16 tile each)
// ---------------------------------------------------------------------------
__global__ __launch_bounds__(128) void esmattn_qkvg_proj(
    const float* __restrict__ q_x, const float* __restrict__ kv_x,
    const _Float16* __restrict__ WT,   // 5 transposed f16 weights
    const float* __restrict__ bg,
    _Float16* __restrict__ qb,   // (B,H,Q,DH), pre-scaled by 1/sqrt(DH)
    _Float16* __restrict__ kb,   // (B,H,K,DH)
    _Float16* __restrict__ vT,   // (B,H,DH,K)
    _Float16* __restrict__ gb)   // (B,Q,HD)  sigmoid(x@Wg + bg)
{
    const int lane = threadIdx.x & 31;
    const int wave = threadIdx.x >> 5;
    const int half = lane >> 4;
    const int l16  = lane & 15;

    const int proj = blockIdx.z;                 // 0=q 1=k 2=v 3=g
    const int r0   = blockIdx.x * 64 + wave * 16;
    const int c0   = blockIdx.y * 16;

    const float* __restrict__ X = (proj == 1 || proj == 2) ? kv_x : q_x;
    const _Float16* __restrict__ Wt = WT + (size_t)proj * CC * CC;

    const int arow = r0 + l16;

    v8f acc = {};
#pragma unroll
    for (int kk = 0; kk < CC; kk += 32) {
        const float* ap = X + (size_t)arow * CC + kk + half * 16;
        v16h a = ldA_f32(ap);
        // contiguous 16 halfs: WT[c0+l16][kk + 16*half .. +15]
        const _Float16* wp = Wt + (size_t)(c0 + l16) * CC + kk + 16 * half;
        v16h b = *(const v16h*)wp;
        acc = __builtin_amdgcn_wmma_f32_16x16x32_f16(false, a, false, b,
                                                     (short)0, acc, false, false);
    }

    const int n = c0 + l16;
    const int h = n >> 5;
    const int d = n & 31;
#pragma unroll
    for (int r = 0; r < 8; ++r) {
        const int gr = r0 + r + 8 * half;
        const int bi = gr >> 10;
        const int si = gr & (QQ - 1);
        float val = acc[r];
        if (proj == 0) {
            val *= 0.17677669529663687f;         // 1/sqrt(32)
            qb[(((size_t)bi * HH + h) * QQ + si) * DH + d] = (_Float16)val;
        } else if (proj == 1) {
            kb[(((size_t)bi * HH + h) * KK + si) * DH + d] = (_Float16)val;
        } else if (proj == 2) {
            vT[(((size_t)bi * HH + h) * DH + d) * KK + si] = (_Float16)val;
        } else {
            float s = val + bg[n];
            s = 1.0f / (1.0f + __expf(-s));
            gb[(size_t)gr * HD + n] = (_Float16)s;
        }
    }
}

// Stage one 16x32 f32 bias_pair tile into LDS: 4 async b128 per lane.
__device__ __forceinline__ void stage_bias(uint32_t lds_base,
                                           const float* __restrict__ gbase,
                                           int lane) {
#pragma unroll
    for (int j = 0; j < 4; ++j) {
        const int cid = lane + 32 * j;       // 0..127 16-byte chunks
        const int row = cid >> 3;            // 0..15
        const int c4  = (cid & 7) * 4;       // float col within row
        async_ld_b128(lds_base + (uint32_t)cid * 16u,
                      gbase + (size_t)row * KK + c4);
    }
}

// ---------------------------------------------------------------------------
// Kernel 2: flash attention with async-staged bias + online softmax + gating.
// grid = (Q/16/4, H, B), block = 128; one 16-row q tile per wave.
// ---------------------------------------------------------------------------
__global__ __launch_bounds__(128) void esmattn_flash(
    const _Float16* __restrict__ qb, const _Float16* __restrict__ kb,
    const _Float16* __restrict__ vT, const _Float16* __restrict__ gb,
    const float* __restrict__ bias_mask,   // (B,1,1,K)
    const float* __restrict__ bias_pair,   // (1,H,Q,K)
    _Float16* __restrict__ ob)             // (B,Q,HD)
{
    __shared__ _Float16 psh[4][16 * 32];        // P staging (per wave)
    __shared__ float    bsh[4][2][16 * 32];     // bias double buffer (per wave)

    const int lane = threadIdx.x & 31;
    const int wave = threadIdx.x >> 5;
    const int half = lane >> 4;
    const int l16  = lane & 15;

    const int b  = blockIdx.z;
    const int h  = blockIdx.y;
    const int q0 = (blockIdx.x * 4 + wave) * 16;
    const size_t bh = (size_t)b * HH + h;

    // q tile as A operand
    const _Float16* qp = qb + (bh * QQ + q0 + l16) * DH + half * 16;
    const v16h qa = cat8(*(const v8h*)qp, *(const v8h*)(qp + 8));

    float mprev[8], lsum[8];
#pragma unroll
    for (int r = 0; r < 8; ++r) { mprev[r] = -1e30f; lsum[r] = 0.0f; }
    v8f o0 = {}, o1 = {};

    _Float16* pw = &psh[wave][0];
    const _Float16* pr = pw + l16 * 32 + half * 16;

    const float* bp_base = bias_pair + ((size_t)h * QQ + q0) * KK;
    const uint32_t lds0 = (uint32_t)(uintptr_t)&bsh[wave][0][0];
    const uint32_t lds1 = (uint32_t)(uintptr_t)&bsh[wave][1][0];

    stage_bias(lds0, bp_base, lane);            // prologue: tile kc=0

    for (int kc = 0; kc < KK; kc += 32) {
        const int cur = (kc >> 5) & 1;
        if (kc + 32 < KK) {
            stage_bias(cur ? lds0 : lds1, bp_base + kc + 32, lane);
            wait_async_le4();                   // older 4 done, newer 4 in flight
        } else {
            wait_async_le0();
        }
        const float* bb = &bsh[wave][cur][0];

        // ---- scores: two 16x16 tiles, reduction = DH = 32 (one WMMA each) ----
        const _Float16* kp0 = kb + (bh * KK + kc + l16) * DH + 16 * half;
        const _Float16* kp1 = kp0 + 16 * DH;
        v8f s0 = __builtin_amdgcn_wmma_f32_16x16x32_f16(
            false, qa, false, *(const v16h*)kp0, (short)0, v8f{}, false, false);
        v8f s1 = __builtin_amdgcn_wmma_f32_16x16x32_f16(
            false, qa, false, *(const v16h*)kp1, (short)0, v8f{}, false, false);

        // ---- biases (mask from global, pair from LDS) ----
        const float bm0 = bias_mask[(size_t)b * KK + kc + l16];
        const float bm1 = bias_mask[(size_t)b * KK + kc + 16 + l16];
#pragma unroll
        for (int r = 0; r < 8; ++r) {
            const int m = r + 8 * half;
            s0[r] += bm0 + bb[m * 32 + l16];
            s1[r] += bm1 + bb[m * 32 + 16 + l16];
        }

        // ---- online softmax (row stats across 16-lane halves) ----
#pragma unroll
        for (int r = 0; r < 8; ++r) {
            float x = fmaxf(s0[r], s1[r]);
            x = fmaxf(x, __shfl_xor(x, 1, 32));
            x = fmaxf(x, __shfl_xor(x, 2, 32));
            x = fmaxf(x, __shfl_xor(x, 4, 32));
            x = fmaxf(x, __shfl_xor(x, 8, 32));
            const float mnew  = fmaxf(mprev[r], x);
            const float scale = __expf(mprev[r] - mnew);
            mprev[r] = mnew;
            const float p0 = __expf(s0[r] - mnew);
            const float p1 = __expf(s1[r] - mnew);
            float ps = p0 + p1;
            ps += __shfl_xor(ps, 1, 32);
            ps += __shfl_xor(ps, 2, 32);
            ps += __shfl_xor(ps, 4, 32);
            ps += __shfl_xor(ps, 8, 32);
            lsum[r] = lsum[r] * scale + ps;
            o0[r] *= scale;
            o1[r] *= scale;
            const int m = r + 8 * half;
            pw[m * 32 + l16]      = (_Float16)p0;
            pw[m * 32 + 16 + l16] = (_Float16)p1;
        }

        // ---- P through LDS -> A-operand layout ----
        wait_ds0();
        const v16h pa = cat8(*(const v8h*)pr, *(const v8h*)(pr + 8));

        // ---- PV: V^T gives contiguous B-operand fragments ----
        const _Float16* vp0 = vT + (bh * DH + l16) * KK + kc + 16 * half;
        const _Float16* vp1 = vp0 + (size_t)16 * KK;
        o0 = __builtin_amdgcn_wmma_f32_16x16x32_f16(
            false, pa, false, *(const v16h*)vp0, (short)0, o0, false, false);
        o1 = __builtin_amdgcn_wmma_f32_16x16x32_f16(
            false, pa, false, *(const v16h*)vp1, (short)0, o1, false, false);
    }

    // ---- normalize, gate, store f16 (B,Q,HD) ----
#pragma unroll
    for (int r = 0; r < 8; ++r) {
        const int m    = r + 8 * half;
        const size_t gr = (size_t)b * QQ + q0 + m;
        const float inv = 1.0f / lsum[r];
        const int c0 = h * DH + l16;
        const float g0 = (float)gb[gr * HD + c0];
        const float g1 = (float)gb[gr * HD + c0 + 16];
        ob[gr * HD + c0]      = (_Float16)(o0[r] * inv * g0);
        ob[gr * HD + c0 + 16] = (_Float16)(o1[r] * inv * g1);
    }
}

// ---------------------------------------------------------------------------
// Kernel 3: out = ob(8192x256 f16) @ Wo(256x256) + bo  -> f32
// grid = (NROW/64, CC/16), block = 128
// ---------------------------------------------------------------------------
__global__ __launch_bounds__(128) void esmattn_out_proj(
    const _Float16* __restrict__ ob, const _Float16* __restrict__ WoT,
    const float* __restrict__ bo, float* __restrict__ out)
{
    const int lane = threadIdx.x & 31;
    const int wave = threadIdx.x >> 5;
    const int half = lane >> 4;
    const int l16  = lane & 15;

    const int r0 = blockIdx.x * 64 + wave * 16;
    const int c0 = blockIdx.y * 16;
    const int arow = r0 + l16;

    v8f acc = {};
#pragma unroll
    for (int kk = 0; kk < HD; kk += 32) {
        const _Float16* ap = ob + (size_t)arow * HD + kk + half * 16;
        v16h a = cat8(*(const v8h*)ap, *(const v8h*)(ap + 8));
        const _Float16* wp = WoT + (size_t)(c0 + l16) * HD + kk + 16 * half;
        v16h b = *(const v16h*)wp;
        acc = __builtin_amdgcn_wmma_f32_16x16x32_f16(false, a, false, b,
                                                     (short)0, acc, false, false);
    }

    const float bias = bo[c0 + l16];
#pragma unroll
    for (int r = 0; r < 8; ++r) {
        const int gr = r0 + r + 8 * half;
        out[(size_t)gr * CC + c0 + l16] = acc[r] + bias;
    }
}

// ---------------------------------------------------------------------------
extern "C" void kernel_launch(void* const* d_in, const int* in_sizes, int n_in,
                              void* d_out, int out_size, void* d_ws, size_t ws_size,
                              hipStream_t stream) {
    const float* q_x       = (const float*)d_in[0];
    const float* kv_x      = (const float*)d_in[1];
    const float* bias_mask = (const float*)d_in[2];
    const float* bias_pair = (const float*)d_in[3];
    const float* Wq        = (const float*)d_in[4];
    const float* Wk        = (const float*)d_in[5];
    const float* Wv        = (const float*)d_in[6];
    const float* Wg        = (const float*)d_in[7];
    const float* bg        = (const float*)d_in[8];
    const float* Wo        = (const float*)d_in[9];
    const float* bo        = (const float*)d_in[10];
    float* out = (float*)d_out;

    const size_t NP = (size_t)NROW * HD;       // 2,097,152 f16 per buffer
    _Float16* ws = (_Float16*)d_ws;
    _Float16* qb = ws;
    _Float16* kb = qb + NP;
    _Float16* vT = kb + NP;
    _Float16* gb = vT + NP;
    _Float16* ob = gb + NP;
    _Float16* WT = ob + NP;                    // 5 x 256 x 256 f16
    _Float16* WoT = WT + (size_t)4 * CC * CC;

    hipLaunchKernelGGL(esmattn_prep_weights, dim3(16, 16, 5), dim3(16, 16),
                       0, stream, Wq, Wk, Wv, Wg, Wo, WT);
    hipLaunchKernelGGL(esmattn_qkvg_proj, dim3(NROW / 64, HD / 16, 4), dim3(128),
                       0, stream, q_x, kv_x, WT, bg, qb, kb, vT, gb);
    hipLaunchKernelGGL(esmattn_flash, dim3(QQ / 64, HH, BB), dim3(128),
                       0, stream, qb, kb, vT, gb, bias_mask, bias_pair, ob);
    hipLaunchKernelGGL(esmattn_out_proj, dim3(NROW / 64, CC / 16, 1), dim3(128),
                       0, stream, ob, WoT, bo, out);
}
